// SSM_55808805044263
// MI455X (gfx1250) — matrix operand order repaired and checked by reading, not compiled, not run on hardware
//
#include <hip/hip_runtime.h>
#include <math.h>

typedef __attribute__((ext_vector_type(16))) _Float16 v16h;
typedef __attribute__((ext_vector_type(8)))  float    v8f;
typedef __attribute__((ext_vector_type(4)))  float    v4f;

#define L_SEQ 16384
#define CHN   2048
#define SDIM  64
#define KCH   64          // CH / 32 k-chunks
#define FRAG_BYTES 2048   // hi(1024) + lo(1024) per 32x16 fragment

// ws layout (bytes)
#define WSWZ_OFF  0u                        // 2*64*4 frags * 2048 B = 1 MB
#define BBUF_OFF  (1u<<20)                  // 16384*64*4 = 4 MB
#define CBUF_OFF  ((1u<<20) + (4u<<20))     // 4 MB
#define DT_OFF    (9u<<20)                  // 64 KB
#define DEX_OFF   ((9u<<20) + 65536u)       // 64 KB
#define BTOT_OFF  ((9u<<20) + 131072u)      // 256 B
#define BPREF_OFF ((9u<<20) + 131328u)      // 256 B

__device__ __forceinline__ v8f wmma16(v16h a, v16h b, v8f c) {
  // D = A(16x32 f16) * B(32x16 f16) + C(16x16 f32)
  return __builtin_amdgcn_wmma_f32_16x16x32_f16(false, a, false, b,
                                                (short)0, c, false, false);
}

// ---------------------------------------------------------------------------
// K0: pre-swizzle W_B / W_C (fp32 [2048,64] row-major) into B-fragment layout,
// split into f16 hi + lo. Fragment f = ((m*64+kc)*4+nt); per-lane 16 halves.
// B 32x16 layout: lanes 0-15 -> N=lane, K={0..7,16..23}; lanes 16-31 -> N=lane-16, K={8..15,24..31}.
// ---------------------------------------------------------------------------
__global__ __launch_bounds__(256)
void swizzle_w_kernel(const float* __restrict__ W_B,
                      const float* __restrict__ W_C,
                      char* __restrict__ wswz) {
  int gid  = blockIdx.x * 256 + threadIdx.x;   // 16384 threads = 512 frags * 32 lanes
  int lane = gid & 31;
  int f    = gid >> 5;
  int nt   = f & 3;
  int kc   = (f >> 2) & 63;
  int m    = f >> 8;
  const float* W = m ? W_C : W_B;
  int ncol = (lane & 15) + nt * 16;
  int kb   = kc * 32 + ((lane >> 4) << 3);     // 0 or 8
  v16h hi, lo;
#pragma unroll
  for (int j = 0; j < 16; ++j) {
    float v = W[(size_t)(kb + (j & 7) + ((j >> 3) << 4)) * SDIM + ncol];
    _Float16 h = (_Float16)v;
    hi[j] = h;
    lo[j] = (_Float16)(v - (float)h);
  }
  char* base = wswz + (size_t)f * FRAG_BYTES + (size_t)lane * 32;
  *(v16h*)base            = hi;
  *(v16h*)(base + 1024)   = lo;
}

// ---------------------------------------------------------------------------
// K1: fused GEMM (B = x@W_B, C = x@W_C via WMMA f16 split-precision) + dt dot.
// Block = 128 threads (4 waves); wave owns 16 rows x 64 cols for both outputs.
// A 16x32 f16 layout: lanes 0-15 row M=lane, K={k0..k0+7, k0+16..k0+23};
//                     lanes 16-31 same rows, K shifted by +8.
// ---------------------------------------------------------------------------
__global__ __launch_bounds__(128)
void gemm_dt_kernel(const float* __restrict__ x,
                    const char*  __restrict__ wswz,
                    const float* __restrict__ W_dt,
                    const float* __restrict__ b_dt,
                    float* __restrict__ Bbuf,
                    float* __restrict__ Cbuf,
                    float* __restrict__ dtb) {
  __shared__ float sdt[CHN];
  const int tid  = threadIdx.x;
  const int lane = tid & 31;
  const int wid  = tid >> 5;

  // stage W_dt into LDS (coalesced)
  for (int i = tid; i < CHN; i += 128) sdt[i] = W_dt[i];
  __syncthreads();

  const int rowBase = blockIdx.x * 64 + wid * 16;
  const float* xrow = x + (size_t)(rowBase + (lane & 15)) * CHN;
  const int ksel = (lane >> 4) << 3;   // 0 or 8

  v8f accB[4] = {};
  v8f accC[4] = {};
  float dtacc = 0.f;

  for (int kc = 0; kc < KCH; ++kc) {
    const int koff = kc * 32 + ksel;
    float xv[16];
    *(float4*)&xv[0]  = *(const float4*)(xrow + koff);
    *(float4*)&xv[4]  = *(const float4*)(xrow + koff + 4);
    *(float4*)&xv[8]  = *(const float4*)(xrow + koff + 16);
    *(float4*)&xv[12] = *(const float4*)(xrow + koff + 20);

    // fused dt partial dot (W_dt from LDS, same K slots this lane owns)
    float wv[16];
    *(float4*)&wv[0]  = *(const float4*)(sdt + koff);
    *(float4*)&wv[4]  = *(const float4*)(sdt + koff + 4);
    *(float4*)&wv[8]  = *(const float4*)(sdt + koff + 16);
    *(float4*)&wv[12] = *(const float4*)(sdt + koff + 20);
#pragma unroll
    for (int j = 0; j < 16; ++j) dtacc = fmaf(xv[j], wv[j], dtacc);

    // split-precision A fragment
    v16h ahi, alo;
#pragma unroll
    for (int j = 0; j < 16; ++j) {
      _Float16 h = (_Float16)xv[j];
      ahi[j] = h;
      alo[j] = (_Float16)(xv[j] - (float)h);
    }

#pragma unroll
    for (int m = 0; m < 2; ++m) {
      v8f* acc = m ? accC : accB;
#pragma unroll
      for (int nt = 0; nt < 4; ++nt) {
        const char* fb = wswz + (size_t)(((m * KCH + kc) * 4) + nt) * FRAG_BYTES
                              + (size_t)lane * 32;
        v16h bhi = *(const v16h*)fb;
        v16h blo = *(const v16h*)(fb + 1024);
        acc[nt] = wmma16(ahi, bhi, acc[nt]);   // hi*hi
        acc[nt] = wmma16(ahi, blo, acc[nt]);   // hi*lo
        acc[nt] = wmma16(alo, bhi, acc[nt]);   // lo*hi
      }
    }
  }

  // finish dt: lane r and lane r+16 hold disjoint K halves of row rowBase+r
  float dot = dtacc + __shfl_xor(dtacc, 16);
  if (lane < 16) {
    float z = dot + b_dt[0];
    float sp = (z > 20.f) ? z : log1pf(expf(z));   // softplus
    dtb[rowBase + lane] = sp;
  }

  // store accumulators: VGPR v -> row rowBase + v + (lane<16?0:8), col = nt*16 + (lane&15)
  const int rb2 = rowBase + ((lane >> 4) << 3);
  const int col = lane & 15;
#pragma unroll
  for (int nt = 0; nt < 4; ++nt) {
#pragma unroll
    for (int v = 0; v < 8; ++v) {
      Bbuf[(size_t)(rb2 + v) * SDIM + nt * 16 + col] = accB[nt][v];
      Cbuf[(size_t)(rb2 + v) * SDIM + nt * 16 + col] = accC[nt][v];
    }
  }
}

// ---------------------------------------------------------------------------
// K2..K4: parallel exclusive prefix sum of dt[16384] (3 tiny stages)
// ---------------------------------------------------------------------------
__global__ __launch_bounds__(256)
void dt_block_sum_kernel(const float* __restrict__ dtb, float* __restrict__ btot) {
  __shared__ float wsum[8];
  int tid = threadIdx.x;
  float v = dtb[blockIdx.x * 256 + tid];
#pragma unroll
  for (int off = 16; off >= 1; off >>= 1) v += __shfl_xor(v, off);
  if ((tid & 31) == 0) wsum[tid >> 5] = v;
  __syncthreads();
  if (tid == 0) {
    float s = 0.f;
#pragma unroll
    for (int i = 0; i < 8; ++i) s += wsum[i];
    btot[blockIdx.x] = s;
  }
}

__global__ __launch_bounds__(64)
void btot_scan_kernel(const float* __restrict__ btot, float* __restrict__ bpref) {
  int i = threadIdx.x;            // 64 threads, exclusive scan of 64 totals
  float s = 0.f;
  for (int j = 0; j < i; ++j) s += btot[j];
  bpref[i] = s;
}

__global__ __launch_bounds__(256)
void dt_exclusive_kernel(const float* __restrict__ dtb,
                         const float* __restrict__ bpref,
                         float* __restrict__ dex) {
  __shared__ float wsum[8];
  int tid = threadIdx.x, lane = tid & 31, wid = tid >> 5;
  int g = blockIdx.x * 256 + tid;
  float v = dtb[g];
  float incl = v;
#pragma unroll
  for (int off = 1; off < 32; off <<= 1) {
    float u = __shfl_up(incl, off);
    if (lane >= off) incl += u;
  }
  if (lane == 31) wsum[wid] = incl;
  __syncthreads();
  float woff = 0.f;
  for (int w = 0; w < wid; ++w) woff += wsum[w];
  dex[g] = bpref[blockIdx.x] + woff + (incl - v);
}

// ---------------------------------------------------------------------------
// K5: g[l] = sum_s C*exp(A*Dexcl)*(exp(A*dt)-1)*B/A ; out[l,:] = g[l]*x[l,:]
// one wave per row; coalesced v4f stream of x -> out (nontemporal stores)
// ---------------------------------------------------------------------------
__global__ __launch_bounds__(256)
void ssm_output_kernel(const float* __restrict__ x,
                       const float* __restrict__ Bbuf,
                       const float* __restrict__ Cbuf,
                       const float* __restrict__ dtb,
                       const float* __restrict__ dex,
                       const float* __restrict__ A,
                       float* __restrict__ out) {
  int tid = threadIdx.x, lane = tid & 31, wid = tid >> 5;
  int l = blockIdx.x * 8 + wid;
  float dtl = dtb[l];
  float Dl  = dex[l];
  float part = 0.f;
#pragma unroll
  for (int k = 0; k < 2; ++k) {
    int s = lane + k * 32;
    float a  = A[s];
    float Bv = Bbuf[(size_t)l * SDIM + s];
    float Cv = Cbuf[(size_t)l * SDIM + s];
    float abar = expf(a * dtl);
    float bbar = (abar - 1.f) * Bv / a;
    part = fmaf(Cv * expf(a * Dl), bbar, part);
  }
#pragma unroll
  for (int off = 16; off >= 1; off >>= 1) part += __shfl_xor(part, off);
  float g = part;

  const v4f* xr = (const v4f*)(x + (size_t)l * CHN);
  v4f* orow = (v4f*)(out + (size_t)l * CHN);
#pragma unroll
  for (int it = 0; it < 16; ++it) {
    int idx = it * 32 + lane;
    v4f v = xr[idx];
    v4f r = v * g;                                  // ext-vector scalar broadcast
    __builtin_nontemporal_store(r, &orow[idx]);
  }
}

// ---------------------------------------------------------------------------
extern "C" void kernel_launch(void* const* d_in, const int* in_sizes, int n_in,
                              void* d_out, int out_size, void* d_ws, size_t ws_size,
                              hipStream_t stream) {
  (void)in_sizes; (void)n_in; (void)out_size; (void)ws_size;
  const float* x    = (const float*)d_in[0];
  const float* W_B  = (const float*)d_in[1];
  const float* W_C  = (const float*)d_in[2];
  const float* W_dt = (const float*)d_in[3];
  const float* b_dt = (const float*)d_in[4];
  const float* A    = (const float*)d_in[5];
  float* out = (float*)d_out;

  char*  ws    = (char*)d_ws;
  char*  wswz  = ws + WSWZ_OFF;
  float* Bbuf  = (float*)(ws + BBUF_OFF);
  float* Cbuf  = (float*)(ws + CBUF_OFF);
  float* dtb   = (float*)(ws + DT_OFF);
  float* dex   = (float*)(ws + DEX_OFF);
  float* btot  = (float*)(ws + BTOT_OFF);
  float* bpref = (float*)(ws + BPREF_OFF);

  // K0: pre-swizzle weights into WMMA B-fragment layout (f16 hi+lo)
  swizzle_w_kernel<<<64, 256, 0, stream>>>(W_B, W_C, wswz);
  // K1: WMMA GEMMs for B and C projections + fused dt dot/softplus
  gemm_dt_kernel<<<L_SEQ / 64, 128, 0, stream>>>(x, wswz, W_dt, b_dt, Bbuf, Cbuf, dtb);
  // K2..K4: exclusive prefix sum of dt
  dt_block_sum_kernel<<<64, 256, 0, stream>>>(dtb, btot);
  btot_scan_kernel<<<1, 64, 0, stream>>>(btot, bpref);
  dt_exclusive_kernel<<<64, 256, 0, stream>>>(dtb, bpref, dex);
  // K5: state contraction + output scaling
  ssm_output_kernel<<<L_SEQ / 8, 256, 0, stream>>>(x, Bbuf, Cbuf, dtb, dex, A, out);
}
